// EfficientRNN_14285061226773
// MI455X (gfx1250) — matrix operand order repaired and checked by reading, not compiled, not run on hardware
//
#include <hip/hip_runtime.h>
#include <hip/hip_bf16.h>

// ---------------------------------------------------------------------------
// EfficientRNN on MI455X (gfx1250).
// Key insight: _gru ignores the hidden state (h_r/h_z/h_n are bias slices), so
// run_stack(x_t, cur) depends only on (x_t, cur), cur in {0,1,2}. We
// precompute all 3 branches for all T steps as 12 large fused GEMM+GRU kernels
// (155 GFLOP of v_wmma_f32_16x16x32_bf16 work, fully parallel), then a single
// persistent workgroup resolves the cheap sequential argmax chain cur[t], and
// a gather kernel emits outputs/h_final from fp32 layer-3 activations.
//
// Workspace budget ~324 MB (hs bf16 201MB + layer3 fp32 100MB + x bf16 17MB
// + bf16 weights 5MB + small buffers).
// ---------------------------------------------------------------------------

#define Bb   128
#define Tt   256
#define Ii   256
#define Hh   256
#define Ll   4
#define Ss   3
#define ROWS (Bb * Tt)            // 32768 GEMM rows (b*T + t)
#define LAYER_ELEMS ((size_t)ROWS * Hh)   // 8,388,608 per layer

typedef __attribute__((ext_vector_type(16))) __bf16 bf16x16;
typedef __attribute__((ext_vector_type(8)))  __bf16 bf16x8;
typedef __attribute__((ext_vector_type(8)))  float  f32x8;

// ---------------------------------------------------------------- fp32->bf16
__global__ void cast_bf16_kernel(const float* __restrict__ src,
                                 __bf16* __restrict__ dst, int n) {
  int i = blockIdx.x * blockDim.x + threadIdx.x;
  if (i < n) dst[i] = (__bf16)src[i];
}

// ------------------------------------------------ fused GEMM + GRU gating.
// Computes h = (1-z)*n with r=sig(x@Wr^T+bi_r+bh_r), z=sig(...), n=tanh(i_n +
// bi_n + r*bh_n). A: ROWSx256 bf16, W: 768x256 bf16 (rows = gate outputs).
// One wave -> one 16x16 output tile, 3 accumulators (gates), K-loop of 8
// steps x 3 WMMAs. 8 waves/block share the same ntile (B-frags hit WGP$).
__global__ __launch_bounds__(256)
void gru_gemm_kernel(const __bf16* __restrict__ A, const __bf16* __restrict__ W,
                     const float* __restrict__ bi, const float* __restrict__ bh,
                     __bf16* __restrict__ Hout, float* __restrict__ Hout32) {
  const int lane  = threadIdx.x & 31;
  const int wave  = threadIdx.x >> 5;
  const int mtile = blockIdx.y * 8 + wave;       // ROWS/16 = 2048 tiles
  const int ntile = blockIdx.x;                  // 16 tiles of H
  const int m0    = mtile * 16;
  const int nl    = lane & 15;                   // A row within tile == C col
  const int half  = lane >> 4;

  const __bf16* Arow = A + (size_t)(m0 + nl) * 256;
  const int ncol = ntile * 16 + nl;
  const __bf16* Wr = W + (size_t)ncol * 256;            // gate r row
  const __bf16* Wz = W + (size_t)(Hh + ncol) * 256;     // gate z row
  const __bf16* Wn = W + (size_t)(2 * Hh + ncol) * 256; // gate n row

  f32x8 accR = {}, accZ = {}, accN = {};
#pragma unroll
  for (int kk = 0; kk < 256; kk += 32) {
    // A-frag: lane (m, half): K chunks [kk+16h, +8) and [kk+16h+8, +8)
    bf16x8 lo = *(const bf16x8*)(Arow + kk + 16 * half);
    bf16x8 hi = *(const bf16x8*)(Arow + kk + 16 * half + 8);
    bf16x16 a;
#pragma unroll
    for (int i = 0; i < 8; ++i) { a[i] = lo[i]; a[8 + i] = hi[i]; }
    // B-frag: lane (n, half): 16 contiguous K at kk+16h from W row n
    bf16x16 br = *(const bf16x16*)(Wr + kk + 16 * half);
    bf16x16 bz = *(const bf16x16*)(Wz + kk + 16 * half);
    bf16x16 bn = *(const bf16x16*)(Wn + kk + 16 * half);
    accR = __builtin_amdgcn_wmma_f32_16x16x32_bf16(false, a, false, br,
                                                   (short)0, accR, false, false);
    accZ = __builtin_amdgcn_wmma_f32_16x16x32_bf16(false, a, false, bz,
                                                   (short)0, accZ, false, false);
    accN = __builtin_amdgcn_wmma_f32_16x16x32_bf16(false, a, false, bn,
                                                   (short)0, accN, false, false);
  }
  const float biR = bi[ncol] + bh[ncol];
  const float biZ = bi[Hh + ncol] + bh[Hh + ncol];
  const float biN = bi[2 * Hh + ncol];
  const float hnN = bh[2 * Hh + ncol];
#pragma unroll
  for (int r = 0; r < 8; ++r) {
    const int mrow = m0 + r + 8 * half;   // C layout: VGPR r -> row r + 8*half
    float rg = 1.f / (1.f + __expf(-(accR[r] + biR)));
    float zg = 1.f / (1.f + __expf(-(accZ[r] + biZ)));
    float ng = tanhf(accN[r] + biN + rg * hnN);
    float v  = (1.f - zg) * ng;
    Hout[(size_t)mrow * Hh + ncol] = (__bf16)v;
    if (Hout32) Hout32[(size_t)mrow * Hh + ncol] = v;  // fp32 copy (layer 3)
  }
}

// ------------------------------------------------ time-invariant folds.
// wcomb[b][l][k] = sum_j hidden[b][l][j] * w_lw[j][k];  cbl = hidden . b_lw
__global__ void wcomb_kernel(const float* __restrict__ hidden,
                             const float* __restrict__ w_lw,
                             const float* __restrict__ b_lw,
                             float* __restrict__ wcomb, float* __restrict__ cbl) {
  int idx = blockIdx.x * blockDim.x + threadIdx.x;       // Bb*Ll*Hh
  if (idx >= Bb * Ll * Hh) return;
  int k = idx & 255, pair = idx >> 8;
  const float* hd = hidden + pair * 4;
  wcomb[idx] = hd[0] * w_lw[k] + hd[1] * w_lw[256 + k] +
               hd[2] * w_lw[512 + k] + hd[3] * w_lw[768 + k];
  if (k == 0)
    cbl[pair] = hd[0] * b_lw[0] + hd[1] * b_lw[1] + hd[2] * b_lw[2] + hd[3] * b_lw[3];
}

// xsum[t][k] = sum_b x[b][t][k]  (batch-reduced input for the score argmax)
__global__ void xsum_kernel(const float* __restrict__ x, float* __restrict__ xsum) {
  int idx = blockIdx.x * blockDim.x + threadIdx.x;       // Tt*Ii
  if (idx >= Tt * Ii) return;
  int t = idx >> 8, k = idx & 255;
  float s = 0.f;
  for (int b = 0; b < Bb; ++b) s += x[((size_t)b * Tt + t) * Ii + k];
  xsum[idx] = s;
}

// ------------------------------------------------ sequential selection chain.
// One persistent 1024-thread workgroup walks t=1..T-1 resolving cur[t].
__global__ __launch_bounds__(1024)
void selection_kernel(const __bf16* __restrict__ hs,    // [S][L][ROWS][H] bf16
                      const float* __restrict__ wcomb,  // [B*L][H]
                      const float* __restrict__ cbl,    // [B*L]
                      const float* __restrict__ w_sel,  // [S][512]
                      const float* __restrict__ b_sel,  // [S]
                      const float* __restrict__ xsum,   // [T][I]
                      int* __restrict__ cur) {
  __shared__ float a_sm[Bb * Ll];
  __shared__ float part[1024];
  __shared__ float ctx[Hh];
  __shared__ float tot[Ss];
  __shared__ int   cur_sm;
  const int tid = threadIdx.x;
  if (tid == 0) { cur_sm = 0; cur[0] = 0; }
  __syncthreads();

  for (int t = 1; t < Tt; ++t) {
    const int sp = cur_sm;  // selection used for h_{t-1}
    const __bf16* hbase = hs + (size_t)sp * Ll * LAYER_ELEMS;
    // --- attention energies: 2 threads per (b,l) pair, split K range
    {
      int pair = tid >> 1, sub = tid & 1;
      int b = pair >> 2, l = pair & 3;
      const __bf16* hrow =
          hbase + (size_t)l * LAYER_ELEMS + ((size_t)b * Tt + (t - 1)) * Hh;
      const float* wc = wcomb + (size_t)pair * Hh;
      float acc = 0.f;
      for (int k = sub * 128; k < sub * 128 + 128; ++k)
        acc += (float)hrow[k] * wc[k];
      part[tid] = acc;
    }
    __syncthreads();
    if (tid < Bb * Ll) a_sm[tid] = part[2 * tid] + part[2 * tid + 1] + cbl[tid];
    __syncthreads();
    // --- softmax over L=4 per batch row
    if (tid < Bb) {
      float e0 = a_sm[4 * tid], e1 = a_sm[4 * tid + 1];
      float e2 = a_sm[4 * tid + 2], e3 = a_sm[4 * tid + 3];
      float m = fmaxf(fmaxf(e0, e1), fmaxf(e2, e3));
      float x0 = __expf(e0 - m), x1 = __expf(e1 - m);
      float x2 = __expf(e2 - m), x3 = __expf(e3 - m);
      float inv = 1.f / (x0 + x1 + x2 + x3);
      a_sm[4 * tid] = x0 * inv; a_sm[4 * tid + 1] = x1 * inv;
      a_sm[4 * tid + 2] = x2 * inv; a_sm[4 * tid + 3] = x3 * inv;
    }
    __syncthreads();
    // --- ctx[k] = sum_b sum_l a[b][l]*h[b][l][k]; 4 threads per k
    {
      int k = tid & 255, grp = tid >> 8;  // grp in 0..3 -> 32 batches each
      float acc = 0.f;
      for (int b = grp * 32; b < grp * 32 + 32; ++b) {
        const __bf16* hb = hbase + ((size_t)b * Tt + (t - 1)) * Hh + k;
        acc += a_sm[4 * b]     * (float)hb[0]
             + a_sm[4 * b + 1] * (float)hb[LAYER_ELEMS]
             + a_sm[4 * b + 2] * (float)hb[2 * LAYER_ELEMS]
             + a_sm[4 * b + 3] * (float)hb[3 * LAYER_ELEMS];
      }
      part[tid] = acc;
    }
    __syncthreads();
    if (tid < Hh) ctx[tid] = part[tid] + part[tid + 256] + part[tid + 512] + part[tid + 768];
    __syncthreads();
    // --- batch-summed scores per s (3 waves, wave32 shuffle reduce)
    if (tid < 96) {
      int s = tid >> 5, lane = tid & 31;
      const float* ws = w_sel + s * 512;
      float p = 0.f;
      for (int k = lane; k < Hh; k += 32)
        p += ctx[k] * ws[k] + xsum[t * Ii + k] * ws[Hh + k];
      for (int off = 16; off; off >>= 1) p += __shfl_down(p, off, 32);
      if (lane == 0) tot[s] = p + (float)Bb * b_sel[s];
    }
    __syncthreads();
    if (tid == 0) {
      int best = 0; float bv = tot[0];
      if (tot[1] > bv) { bv = tot[1]; best = 1; }
      if (tot[2] > bv) { bv = tot[2]; best = 2; }
      cur_sm = best; cur[t] = best;
    }
    __syncthreads();
  }
}

// ------------------------------------------------ gather outputs + h_final
__global__ void gather_kernel(const float* __restrict__ hs3,  // [S][ROWS][H] f32
                              const int* __restrict__ cur, float* __restrict__ out) {
  size_t idx = (size_t)blockIdx.x * blockDim.x + threadIdx.x;
  const size_t N = (size_t)Bb * Tt * Hh;
  if (idx < N) {
    size_t k = idx & 255, t = (idx >> 8) & 255, b = idx >> 16;
    int s = cur[t];
    out[idx] = hs3[((size_t)s * ROWS + b * Tt + t) * Hh + k];
  } else if (idx < N + (size_t)Bb * Hh) {
    size_t r = idx - N, k = r & 255, b = r >> 8;
    int s = cur[Tt - 1];
    out[idx] = hs3[((size_t)s * ROWS + b * Tt + (Tt - 1)) * Hh + k];
  }
}

// ---------------------------------------------------------------------------
extern "C" void kernel_launch(void* const* d_in, const int* in_sizes, int n_in,
                              void* d_out, int out_size, void* d_ws, size_t ws_size,
                              hipStream_t stream) {
  const float* x      = (const float*)d_in[0];
  const float* hidden = (const float*)d_in[1];
  const float* w_ih0  = (const float*)d_in[2];
  const float* b_ih0  = (const float*)d_in[3];
  const float* b_hh0  = (const float*)d_in[4];
  const float* w_ih   = (const float*)d_in[5];
  const float* b_ih   = (const float*)d_in[6];
  const float* b_hh   = (const float*)d_in[7];
  const float* w_lw   = (const float*)d_in[8];
  const float* b_lw   = (const float*)d_in[9];
  const float* w_sel  = (const float*)d_in[10];
  const float* b_sel  = (const float*)d_in[11];
  float* out = (float*)d_out;

  // workspace carve-up (256B aligned regions)
  char* p = (char*)d_ws;
  auto carve = [&](size_t bytes) -> char* {
    char* r = p; p += (bytes + 255) & ~(size_t)255; return r;
  };
  const size_t NX  = (size_t)Bb * Tt * Ii;          // 8,388,608
  const size_t NW0 = (size_t)Ss * 3 * Hh * Ii;      //   589,824
  const size_t NW  = (size_t)Ss * (Ll - 1) * 3 * Hh * Hh; // 1,769,472
  __bf16* xb   = (__bf16*)carve(NX * 2);
  __bf16* w0b  = (__bf16*)carve(NW0 * 2);
  __bf16* wb   = (__bf16*)carve(NW * 2);
  __bf16* hs   = (__bf16*)carve((size_t)Ss * Ll * LAYER_ELEMS * 2); // 201 MB
  float*  hs3  = (float*)carve((size_t)Ss * LAYER_ELEMS * 4);       // 100 MB
  float*  xsum = (float*)carve((size_t)Tt * Ii * 4);
  float*  wcmb = (float*)carve((size_t)Bb * Ll * Hh * 4);
  float*  cbl  = (float*)carve((size_t)Bb * Ll * 4);
  int*    cur  = (int*)carve((size_t)Tt * 4);

  // 1) bf16 conversions
  cast_bf16_kernel<<<(int)((NX  + 255) / 256), 256, 0, stream>>>(x, xb, (int)NX);
  cast_bf16_kernel<<<(int)((NW0 + 255) / 256), 256, 0, stream>>>(w_ih0, w0b, (int)NW0);
  cast_bf16_kernel<<<(int)((NW  + 255) / 256), 256, 0, stream>>>(w_ih, wb, (int)NW);

  // 2) time-invariant folds (independent of GEMMs)
  wcomb_kernel<<<(Bb * Ll * Hh + 255) / 256, 256, 0, stream>>>(hidden, w_lw, b_lw, wcmb, cbl);
  xsum_kernel<<<(Tt * Ii + 255) / 256, 256, 0, stream>>>(x, xsum);

  // 3) 12 fused GEMM+GRU passes: all (s, layer) branches over all T steps
  dim3 ggrid(Hh / 16, (ROWS / 16) / 8);   // 16 x 256 blocks
  dim3 gblock(256);                       // 8 waves, one 16x16 tile each
  for (int s = 0; s < Ss; ++s) {
    // layer 0: A = x (bf16)
    gru_gemm_kernel<<<ggrid, gblock, 0, stream>>>(
        xb, w0b + (size_t)s * 3 * Hh * Ii,
        b_ih0 + (size_t)s * 3 * Hh, b_hh0 + (size_t)s * 3 * Hh,
        hs + (size_t)(s * Ll + 0) * LAYER_ELEMS, (float*)nullptr);
    // layers 1..3: A = previous layer output
    for (int l = 1; l < Ll; ++l) {
      const int wl = s * (Ll - 1) + (l - 1);
      gru_gemm_kernel<<<ggrid, gblock, 0, stream>>>(
          hs + (size_t)(s * Ll + l - 1) * LAYER_ELEMS,
          wb + (size_t)wl * 3 * Hh * Hh,
          b_ih + (size_t)wl * 3 * Hh, b_hh + (size_t)wl * 3 * Hh,
          hs + (size_t)(s * Ll + l) * LAYER_ELEMS,
          (l == Ll - 1) ? (hs3 + (size_t)s * LAYER_ELEMS) : (float*)nullptr);
    }
  }

  // 4) sequential argmax chain (single persistent workgroup)
  selection_kernel<<<1, 1024, 0, stream>>>(hs, wcmb, cbl, w_sel, b_sel, xsum, cur);

  // 5) assemble outputs (B,T,H) + h_final (B,H)
  const size_t NOUT = (size_t)Bb * Tt * Hh + (size_t)Bb * Hh;
  gather_kernel<<<(int)((NOUT + 255) / 256), 256, 0, stream>>>(hs3, cur, out);
}
